// RingAttentionLayer_37374805409934
// MI455X (gfx1250) — matrix operand order repaired and checked by reading, not compiled, not run on hardware
//
#include <hip/hip_runtime.h>
#include <hip/hip_bf16.h>
#include <math.h>

typedef __attribute__((ext_vector_type(16))) __bf16 v16bf;
typedef __attribute__((ext_vector_type(8)))  __bf16 v8bf;
typedef __attribute__((ext_vector_type(4)))  __bf16 v4bf;
typedef __attribute__((ext_vector_type(8)))  float  v8f;

#define SEQ  2048
#define HID  2048
#define NH   16
#define HD   128

static __device__ __forceinline__ v16bf pack16(v8bf lo, v8bf hi) {
  v16bf r;
#pragma unroll
  for (int i = 0; i < 8; ++i) { r[i] = lo[i]; r[i + 8] = hi[i]; }
  return r;
}

// ---------------------------------------------------------------------------
// C = (A * B^T + bias) * outScale    (M=N=K=2048, all row-major)
// A: f32 or bf16 [M][K]; B: f32 [N][K]; C: bf16 or f32 [M][N]
// Block: 256 thr (8 waves, 2x4), tile 128x128, K-step 32, bf16 WMMA.
// ---------------------------------------------------------------------------
template<bool A_BF16, bool OUT_BF16>
__global__ __launch_bounds__(256)
void gemm_nt_bias(const void* __restrict__ Aptr,
                  const float* __restrict__ Bw,
                  const float* __restrict__ bias,
                  void* __restrict__ Cptr,
                  float outScale)
{
  __shared__ __bf16 As[128 * 40];   // pitch 40 elems = 80B (16B aligned)
  __shared__ __bf16 Bs[128 * 40];

  const int tid  = threadIdx.x;
  const int lane = tid & 31;
  const int wave = tid >> 5;
  const int wm   = wave >> 2;       // 0..1 : 64 rows each
  const int wn   = wave & 3;        // 0..3 : 32 cols each
  const int lh   = lane >> 4;       // lane half
  const int l16  = lane & 15;
  const int mblk = blockIdx.y * 128;
  const int nblk = blockIdx.x * 128;

  v8f acc[4][2];
  {
    v8f z = {};
#pragma unroll
    for (int i = 0; i < 4; ++i)
#pragma unroll
      for (int j = 0; j < 2; ++j) acc[i][j] = z;
  }

  const int r0 = tid >> 3;          // 0..31
  const int c0 = (tid & 7) * 4;     // 0..28

  for (int k0 = 0; k0 < HID; k0 += 32) {
    __syncthreads();
    // stage A tile (128x32) and B tile (128 weight rows x 32) as bf16,
    // packed 4-wide so LDS stores become ds_store_b64
#pragma unroll
    for (int rr = 0; rr < 4; ++rr) {
      int r = r0 + rr * 32;
      v4bf a4;
      if constexpr (A_BF16) {
        a4 = *(const v4bf*)((const __bf16*)Aptr + (size_t)(mblk + r) * HID + k0 + c0);
      } else {
        float4 f = *(const float4*)((const float*)Aptr + (size_t)(mblk + r) * HID + k0 + c0);
        a4[0] = (__bf16)f.x; a4[1] = (__bf16)f.y; a4[2] = (__bf16)f.z; a4[3] = (__bf16)f.w;
      }
      *(v4bf*)&As[r * 40 + c0] = a4;

      float4 g = *(const float4*)(Bw + (size_t)(nblk + r) * HID + k0 + c0);
      v4bf b4;
      b4[0] = (__bf16)g.x; b4[1] = (__bf16)g.y; b4[2] = (__bf16)g.z; b4[3] = (__bf16)g.w;
      *(v4bf*)&Bs[r * 40 + c0] = b4;
    }
    __syncthreads();

    v16bf af[4], bfr[2];
#pragma unroll
    for (int i = 0; i < 4; ++i) {
      // A frag 16x32: lane holds row M=l16; K = lh*8 + {0..7, 16..23}
      const __bf16* p = &As[(wm * 64 + i * 16 + l16) * 40 + lh * 8];
      af[i] = pack16(*(const v8bf*)p, *(const v8bf*)(p + 16));
    }
#pragma unroll
    for (int j = 0; j < 2; ++j) {
      // B frag 32x16: lane holds col N=l16; K = lh*8 + {0..7, 16..23}
      const __bf16* p = &Bs[(wn * 32 + j * 16 + l16) * 40 + lh * 8];
      bfr[j] = pack16(*(const v8bf*)p, *(const v8bf*)(p + 16));
    }
#pragma unroll
    for (int i = 0; i < 4; ++i)
#pragma unroll
      for (int j = 0; j < 2; ++j)
        acc[i][j] = __builtin_amdgcn_wmma_f32_16x16x32_bf16(
            false, af[i], false, bfr[j], (short)0, acc[i][j], false, false);
  }

  // epilogue: C layout row = base + e + 8*lh, col = base + l16
#pragma unroll
  for (int i = 0; i < 4; ++i) {
    int rbase = mblk + wm * 64 + i * 16 + lh * 8;
#pragma unroll
    for (int j = 0; j < 2; ++j) {
      int col = nblk + wn * 32 + j * 16 + l16;
      float bv = bias[col];
#pragma unroll
      for (int e = 0; e < 8; ++e) {
        float v = (acc[i][j][e] + bv) * outScale;
        size_t idx = (size_t)(rbase + e) * HID + col;
        if constexpr (OUT_BF16) ((__bf16*)Cptr)[idx] = (__bf16)v;
        else                    ((float*)Cptr)[idx]  = v;
      }
    }
  }
}

// ---------------------------------------------------------------------------
// Flash attention with ring-collapse bias: weight = ws*exp(s) for k<=q,
// (ws-1)*exp(s) for k>q  ==  softmax(s + ln(ws or ws-1)).
// Scale 1/sqrt(D) is pre-folded into Q. Block: 128 thr = 4 waves,
// each wave owns 16 query rows; KV staged in 64-key blocks.
// K tile staged via gfx1250 async global->LDS DMA (ASYNCcnt path);
// V tile transposed through VGPRs into LDS.
// ---------------------------------------------------------------------------
__global__ __launch_bounds__(128)
void flash_attn_ring(const __bf16* __restrict__ Qm,
                     const __bf16* __restrict__ Km,
                     const __bf16* __restrict__ Vm,
                     __bf16* __restrict__ Om,
                     const int* __restrict__ wsz_p)
{
  __shared__ __bf16 Ks[64 * 136];     // [key][dim], pitch 272B
  __shared__ __bf16 Vt[128 * 72];     // [dim][key], pitch 144B
  __shared__ __bf16 Pb[4][16 * 72];   // per-wave P tile, pitch 144B

  const int tid  = threadIdx.x;
  const int lane = tid & 31, wave = tid >> 5;
  const int lh   = lane >> 4, l16 = lane & 15;
  const int h    = blockIdx.y;
  const int qb   = blockIdx.x * 64 + wave * 16;

  const int   ws    = wsz_p[0];
  const float lnIn  = __logf((float)ws);
  const float lnOut = __logf((float)(ws - 1));

  // Q fragments (16 rows x 128 dims = 4 frags of 16x32), held in VGPRs
  v16bf qf[4];
#pragma unroll
  for (int dc = 0; dc < 4; ++dc) {
    const __bf16* s = Qm + (size_t)(qb + l16) * HID + h * HD + dc * 32 + lh * 8;
    qf[dc] = pack16(*(const v8bf*)s, *(const v8bf*)(s + 16));
  }

  v8f acc[8];
  {
    v8f z = {};
#pragma unroll
    for (int d = 0; d < 8; ++d) acc[d] = z;
  }
  float m8[8], l8[8];
#pragma unroll
  for (int e = 0; e < 8; ++e) { m8[e] = -1e30f; l8[e] = 0.f; }

  const int stg_r  = tid >> 1;            // key row 0..63
  const int stg_ch = (tid & 1) * 64;      // dim half

  for (int kb = 0; kb < SEQ; kb += 64) {
    __syncthreads();
    {
      // --- K tile: async DMA global -> LDS (16B per lane per op) ---
      const __bf16* ksrc = Km + (size_t)(kb + stg_r) * HID + h * HD + stg_ch;
      // low 32 bits of a generic shared pointer == LDS byte offset
      unsigned ldsK = (unsigned)(size_t)&Ks[stg_r * 136 + stg_ch];
#pragma unroll
      for (int j = 0; j < 64; j += 8) {
        unsigned long long ga = (unsigned long long)(size_t)(ksrc + j);
        unsigned la = ldsK + (unsigned)(j * 2);
        asm volatile("global_load_async_to_lds_b128 %0, %1, off"
                     :: "v"(la), "v"(ga) : "memory");
      }
      // --- V tile: load through VGPRs, transpose into LDS ---
      const __bf16* vsrc = Vm + (size_t)(kb + stg_r) * HID + h * HD + stg_ch;
#pragma unroll
      for (int j = 0; j < 64; j += 8) {
        v8bf vv = *(const v8bf*)(vsrc + j);
#pragma unroll
        for (int e = 0; e < 8; ++e) Vt[(stg_ch + j + e) * 72 + stg_r] = vv[e];
      }
      if (kb + 64 < SEQ) {                // global_prefetch_b8 of next V block
        __builtin_prefetch(Vm + (size_t)(kb + 64 + stg_r) * HID + h * HD + stg_ch, 0, 1);
      }
    }
    asm volatile("s_wait_asynccnt 0x0" ::: "memory");
    __syncthreads();

    // S = Q * K^T : 4 key tiles of 16, each accumulated over 4 dim chunks
    v8f sacc[4];
    {
      v8f z = {};
#pragma unroll
      for (int nt = 0; nt < 4; ++nt) sacc[nt] = z;
    }
#pragma unroll
    for (int nt = 0; nt < 4; ++nt)
#pragma unroll
      for (int dc = 0; dc < 4; ++dc) {
        const __bf16* p = &Ks[(nt * 16 + l16) * 136 + dc * 32 + lh * 8];
        v16bf kf = pack16(*(const v8bf*)p, *(const v8bf*)(p + 16));
        sacc[nt] = __builtin_amdgcn_wmma_f32_16x16x32_bf16(
            false, qf[dc], false, kf, (short)0, sacc[nt], false, false);
      }

    // add ring bias, block row-max (rows live in 16-lane halves)
    float mb[8];
#pragma unroll
    for (int e = 0; e < 8; ++e) mb[e] = -1e30f;
#pragma unroll
    for (int nt = 0; nt < 4; ++nt)
#pragma unroll
      for (int e = 0; e < 8; ++e) {
        int kg = kb + nt * 16 + l16;
        int qg = qb + e + lh * 8;
        float s = sacc[nt][e] + ((kg <= qg) ? lnIn : lnOut);
        sacc[nt][e] = s;
        mb[e] = fmaxf(mb[e], s);
      }
#pragma unroll
    for (int e = 0; e < 8; ++e) {
      mb[e] = fmaxf(mb[e], __shfl_xor(mb[e], 1, 32));
      mb[e] = fmaxf(mb[e], __shfl_xor(mb[e], 2, 32));
      mb[e] = fmaxf(mb[e], __shfl_xor(mb[e], 4, 32));
      mb[e] = fmaxf(mb[e], __shfl_xor(mb[e], 8, 32));
    }

    float so[8];
#pragma unroll
    for (int e = 0; e < 8; ++e) {
      float mn = fmaxf(m8[e], mb[e]);
      so[e] = __expf(m8[e] - mn);
      m8[e] = mn;
      l8[e] *= so[e];
    }
#pragma unroll
    for (int d = 0; d < 8; ++d)
#pragma unroll
      for (int e = 0; e < 8; ++e) acc[d][e] *= so[e];

    float rs[8];
#pragma unroll
    for (int e = 0; e < 8; ++e) rs[e] = 0.f;
#pragma unroll
    for (int nt = 0; nt < 4; ++nt)
#pragma unroll
      for (int e = 0; e < 8; ++e) {
        float p = __expf(sacc[nt][e] - m8[e]);
        sacc[nt][e] = p;
        rs[e] += p;
      }
#pragma unroll
    for (int e = 0; e < 8; ++e) {
      rs[e] += __shfl_xor(rs[e], 1, 32);
      rs[e] += __shfl_xor(rs[e], 2, 32);
      rs[e] += __shfl_xor(rs[e], 4, 32);
      rs[e] += __shfl_xor(rs[e], 8, 32);
      l8[e] += rs[e];
    }

    // P (C-layout) -> LDS -> A-fragment layout
#pragma unroll
    for (int nt = 0; nt < 4; ++nt)
#pragma unroll
      for (int e = 0; e < 8; ++e)
        Pb[wave][(e + lh * 8) * 72 + nt * 16 + l16] = (__bf16)sacc[nt][e];
    __syncthreads();

    // O += P * V  (two K-chunks of 32 keys, 8 dim tiles of 16)
#pragma unroll
    for (int kh = 0; kh < 2; ++kh) {
      const __bf16* ap = &Pb[wave][l16 * 72 + kh * 32 + lh * 8];
      v16bf af = pack16(*(const v8bf*)ap, *(const v8bf*)(ap + 16));
#pragma unroll
      for (int d = 0; d < 8; ++d) {
        const __bf16* vp = &Vt[(d * 16 + l16) * 72 + kh * 32 + lh * 8];
        v16bf vf = pack16(*(const v8bf*)vp, *(const v8bf*)(vp + 16));
        acc[d] = __builtin_amdgcn_wmma_f32_16x16x32_bf16(
            false, af, false, vf, (short)0, acc[d], false, false);
      }
    }
  }

  // finalize: divide by softmax sum, store bf16 into [S][H*D]
#pragma unroll
  for (int d = 0; d < 8; ++d)
#pragma unroll
    for (int e = 0; e < 8; ++e) {
      float v = acc[d][e] / (l8[e] + 1e-8f);
      Om[(size_t)(qb + e + lh * 8) * HID + h * HD + d * 16 + l16] = (__bf16)v;
    }
}

// ---------------------------------------------------------------------------
extern "C" void kernel_launch(void* const* d_in, const int* in_sizes, int n_in,
                              void* d_out, int out_size, void* d_ws, size_t ws_size,
                              hipStream_t stream)
{
  const float* X   = (const float*)d_in[0];
  const float* q_w = (const float*)d_in[1];
  const float* q_b = (const float*)d_in[2];
  const float* k_w = (const float*)d_in[3];
  const float* k_b = (const float*)d_in[4];
  const float* v_w = (const float*)d_in[5];
  const float* v_b = (const float*)d_in[6];
  const float* o_w = (const float*)d_in[7];
  const float* o_b = (const float*)d_in[8];
  const int*   wsz = (const int*)d_in[9];
  // d_in[10] (local_block_idx) unused: result is independent of it.

  char* ws = (char*)d_ws;
  __bf16* Qb = (__bf16*)(ws);
  __bf16* Kb = (__bf16*)(ws + (size_t)SEQ * HID * 2);
  __bf16* Vb = (__bf16*)(ws + (size_t)SEQ * HID * 4);
  __bf16* Ab = (__bf16*)(ws + (size_t)SEQ * HID * 6);

  dim3 gg(HID / 128, SEQ / 128);
  const float qscale = 0.08838834764831845f;  // 1/sqrt(128), folded into Q

  gemm_nt_bias<false, true><<<gg, 256, 0, stream>>>(X, q_w, q_b, Qb, qscale);
  gemm_nt_bias<false, true><<<gg, 256, 0, stream>>>(X, k_w, k_b, Kb, 1.0f);
  gemm_nt_bias<false, true><<<gg, 256, 0, stream>>>(X, v_w, v_b, Vb, 1.0f);

  dim3 ga(SEQ / 64, NH);
  flash_attn_ring<<<ga, 128, 0, stream>>>(Qb, Kb, Vb, Ab, wsz);

  gemm_nt_bias<true, false><<<gg, 256, 0, stream>>>(Ab, o_w, o_b, (float*)d_out, 1.0f);
}
